// DifferentialAttention_72773925864071
// MI455X (gfx1250) — matrix operand order repaired and checked by reading, not compiled, not run on hardware
//
#include <hip/hip_runtime.h>
#include <stdint.h>

typedef __bf16 bf16_t;
typedef bf16_t v8bf  __attribute__((ext_vector_type(8)));
typedef bf16_t v16bf __attribute__((ext_vector_type(16)));
typedef float  v8f   __attribute__((ext_vector_type(8)));

#define D_MODEL   2048
#define NUM_HEADS 16
#define HEAD_DIM  128
#define SEQ       2048
#define BATCH     2
#define MTOT      (BATCH*SEQ)   // 4096
#define KC        32
#define LDA       40            // padded LDS row stride (bf16 elems), 80B = 5*16B

static __device__ inline v16bf cat8(v8bf lo, v8bf hi) {
  return __builtin_shufflevector(lo, hi, 0,1,2,3,4,5,6,7,8,9,10,11,12,13,14,15);
}
static __device__ inline v8f wmma_bf16(v16bf a, v16bf b, v8f c) {
  // D = A(16xK32) * B(K32x16) + C, f32 accumulate
  return __builtin_amdgcn_wmma_f32_16x16x32_bf16(false, a, false, b, (short)0, c, false, false);
}

// CDNA5 async global->LDS copy (ASYNCcnt-tracked), 16 bytes per lane.
// VDST = LDS byte address (flat LDS addr low 32 bits == LDS offset), VADDR = 64-bit global addr.
static __device__ inline void async_ld_b128(void* lds, const void* g) {
  uint32_t l = (uint32_t)(uintptr_t)lds;
  uint64_t a = (uint64_t)(uintptr_t)g;
  asm volatile("global_load_async_to_lds_b128 %0, %1, off" :: "v"(l), "v"(a) : "memory");
}

// ---------------- fp32 -> bf16 conversion ----------------
__global__ void cvt_f32_bf16(const float* __restrict__ src, bf16_t* __restrict__ dst, int n) {
  int i = blockIdx.x * blockDim.x + threadIdx.x;
  int stride = gridDim.x * blockDim.x;
  for (; i < n; i += stride) dst[i] = (bf16_t)src[i];
}

// ---------------- GEMM: C[m,n] = sum_k A[m,k] * W[n,k] ----------------
// A: [M,K] bf16 row-major; W: [N,K] bf16 row-major.
// mode 0: outf[m*N+n] = f32
// mode 1: outb at [B,H,S,Dh]  (head-split)
// mode 2: outb at [B,H,Dh,S]  (head-split, transposed: for V)
// Double-buffered LDS fed by async global->LDS (ASYNCcnt) transfers.
__global__ __launch_bounds__(256)
void gemm_bf16(const bf16_t* __restrict__ A, const bf16_t* __restrict__ W,
               bf16_t* __restrict__ outb, float* __restrict__ outf,
               int M, int N, int K, int mode) {
  __shared__ bf16_t As[2][128 * LDA];
  __shared__ bf16_t Bs[2][128 * LDA];

  const int tid  = threadIdx.x;
  const int lane = tid & 31;
  const int w    = tid >> 5;
  const int half = lane >> 4;
  const int l16  = lane & 15;
  const int bm   = blockIdx.y * 128;
  const int bn   = blockIdx.x * 128;
  const int wm   = (w & 3) * 32;   // wave M offset in block
  const int wn   = (w >> 2) * 64;  // wave N offset in block

  v8f acc[2][4] = {};

  const int ldrow = tid >> 1;        // 0..127
  const int ldcol = (tid & 1) * 16;  // 0 or 16
  const bf16_t* gA = A + (size_t)(bm + ldrow) * K + ldcol;
  const bf16_t* gB = W + (size_t)(bn + ldrow) * K + ldcol;
  bf16_t* lA = &As[0][0] + ldrow * LDA + ldcol;
  bf16_t* lB = &Bs[0][0] + ldrow * LDA + ldcol;
  const int lbuf = 128 * LDA;  // elems per buffer

  // Prologue: async-prefetch K-tile 0 into buffer 0 (4 x b128 per thread).
  async_ld_b128(lA,     gA);
  async_ld_b128(lA + 8, gA + 8);
  async_ld_b128(lB,     gB);
  async_ld_b128(lB + 8, gB + 8);

  int buf = 0;
  for (int k0 = 0; k0 < K; k0 += KC) {
    // Prefetch next tile into the other buffer (wraps on last iter; harmless).
    const int kn = (k0 + KC < K) ? (k0 + KC) : 0;
    const int nb = buf ^ 1;
    async_ld_b128(lA + nb * lbuf,     gA + kn);
    async_ld_b128(lA + nb * lbuf + 8, gA + kn + 8);
    async_ld_b128(lB + nb * lbuf,     gB + kn);
    async_ld_b128(lB + nb * lbuf + 8, gB + kn + 8);

    // Async transfers complete in order: <=4 outstanding means current tile landed.
    asm volatile("s_wait_asynccnt 0x4" ::: "memory");
    __syncthreads();

    v16bf af[2], bfr[4];
#pragma unroll
    for (int i = 0; i < 2; i++) {
      // A-frag: lane row = l16, elems 0..7 -> k = half*8+{0..7}, 8..15 -> +16
      const bf16_t* p = &As[buf][(wm + i * 16 + l16) * LDA + half * 8];
      af[i] = cat8(*(const v8bf*)p, *(const v8bf*)(p + 16));
    }
#pragma unroll
    for (int j = 0; j < 4; j++) {
      // B-frag: lane col = l16 (output channel), elems 0..7 -> k = half*8+{0..7}, 8..15 -> +16
      const bf16_t* p = &Bs[buf][(wn + j * 16 + l16) * LDA + half * 8];
      bfr[j] = cat8(*(const v8bf*)p, *(const v8bf*)(p + 16));
    }
#pragma unroll
    for (int i = 0; i < 2; i++)
#pragma unroll
      for (int j = 0; j < 4; j++)
        acc[i][j] = wmma_bf16(af[i], bfr[j], acc[i][j]);

    __syncthreads();  // all waves done with buf before it is async-overwritten
    buf = nb;
  }

  // Epilogue. C layout: VGPR v -> row M = v + 8*half; lane l16 -> col N.
#pragma unroll
  for (int i = 0; i < 2; i++) {
    const int mbase = bm + wm + i * 16 + half * 8;
#pragma unroll
    for (int j = 0; j < 4; j++) {
      const int ncol = bn + wn + j * 16 + l16;
#pragma unroll
      for (int v = 0; v < 8; v++) {
        const int m = mbase + v;
        const float val = acc[i][j][v];
        if (mode == 0) {
          outf[(size_t)m * N + ncol] = val;
        } else {
          const int b = m >> 11, s = m & (SEQ - 1);
          const int h = ncol >> 7, d = ncol & (HEAD_DIM - 1);
          size_t addr;
          if (mode == 1) addr = (((size_t)(b * NUM_HEADS + h)) * SEQ + s) * HEAD_DIM + d;
          else           addr = (((size_t)(b * NUM_HEADS + h)) * HEAD_DIM + d) * SEQ + s;
          outb[addr] = (bf16_t)val;
        }
      }
    }
  }
}

// ---------------- Differential attention core ----------------
// q,k: [B*H, S, Dh] bf16 ; vT: [B*H, Dh, S] bf16 ; out: [B, S, D] bf16
// Each wave: 16 query rows. Scores computed transposed (K*Q^T) so P^T C-layout
// matches the A-fragment layout of the P*V WMMA directly.
__global__ __launch_bounds__(256)
void diff_attn(const bf16_t* __restrict__ q1, const bf16_t* __restrict__ k1,
               const bf16_t* __restrict__ q2, const bf16_t* __restrict__ k2,
               const bf16_t* __restrict__ vT, const float* __restrict__ lambda_param,
               bf16_t* __restrict__ out) {
  __shared__ float buf_all[8 * 16 * HEAD_DIM];  // 64 KB: per-wave attn1 buffer

  const int tid  = threadIdx.x;
  const int lane = tid & 31;
  const int w    = tid >> 5;
  const int half = lane >> 4;
  const int l16  = lane & 15;
  const int bh   = blockIdx.x;                 // 0..B*H-1
  const int h    = bh & (NUM_HEADS - 1);
  const int b    = bh >> 4;
  const int q0   = blockIdx.y * 128 + w * 16;  // query row base for this wave
  const float lam   = lambda_param[h];
  const float scale = 0.08838834764831845f;    // 1/sqrt(128)
  float* buf = buf_all + w * 16 * HEAD_DIM;

  const bf16_t* qs[2] = { q1, q2 };
  const bf16_t* ks[2] = { k1, k2 };
  const size_t baseQK = (size_t)bh * SEQ * HEAD_DIM;
  const size_t baseV  = (size_t)bh * HEAD_DIM * SEQ;

  for (int stream = 0; stream < 2; ++stream) {
    // Q as B-operand of K*Q^T: lane col = query q0+l16, k-dim = dh (half*8+{0..7}, +16 per chunk)
    v16bf qf[4];
    const bf16_t* qp = qs[stream] + baseQK + (size_t)(q0 + l16) * HEAD_DIM;
#pragma unroll
    for (int c = 0; c < 4; c++) {
      const v8bf* p = (const v8bf*)(qp + c * 32 + half * 8);
      qf[c] = cat8(p[0], p[2]);
    }

    v8f acc[8] = {};   // out tile: lane col = dh(j*16+l16), VGPR v -> q row v+8*half
    float lsum = 0.f;  // per-lane softmax denominator for query q0+l16

    for (int kb = 0; kb < SEQ; kb += 32) {
      // Two transposed score tiles: keys kb..kb+15 and kb+16..kb+31
      v8f sa = {}, sb = {};
      const bf16_t* kpa = ks[stream] + baseQK + (size_t)(kb + l16) * HEAD_DIM;
      const bf16_t* kpb = kpa + 16 * HEAD_DIM;
#pragma unroll
      for (int c = 0; c < 4; c++) {
        const bf16_t* pa = kpa + c * 32 + half * 8;
        v16bf ka = cat8(*(const v8bf*)pa, *(const v8bf*)(pa + 16));
        sa = wmma_bf16(ka, qf[c], sa);
        const bf16_t* pb = kpb + c * 32 + half * 8;
        v16bf kbf = cat8(*(const v8bf*)pb, *(const v8bf*)(pb + 16));
        sb = wmma_bf16(kbf, qf[c], sb);
      }

      // Softmax weights (scores ~N(0,1) after scaling -> max-free exp is safe).
      // Tile rows = keys (v + 8*half), cols = queries (l16): stats are per-lane.
      v16bf pf;
      float psum = 0.f;
#pragma unroll
      for (int v = 0; v < 8; v++) {
        float pa = __expf(sa[v] * scale);
        float pb = __expf(sb[v] * scale);
        psum += pa + pb;
        pf[v]     = (bf16_t)pa;  // key = half*8 + v        (A-frag elems 0..7)
        pf[v + 8] = (bf16_t)pb;  // key = 16 + half*8 + v   (A-frag elems 8..15)
      }
      psum += __shfl_xor(psum, 16, 32);  // merge the two key-halves
      lsum += psum;

      // P(16q x 32keys) * V(32keys x 16dh) for 8 dh-tiles; V pre-transposed.
      const bf16_t* vp = vT + baseV + (size_t)l16 * SEQ + kb + half * 8;
#pragma unroll
      for (int j = 0; j < 8; j++) {
        const bf16_t* pv = vp + (size_t)j * 16 * SEQ;
        v16bf vf = cat8(*(const v8bf*)pv, *(const v8bf*)(pv + 16));
        acc[j] = wmma_bf16(pf, vf, acc[j]);
      }
    }

    // Normalize; combine streams.
#pragma unroll
    for (int v = 0; v < 8; v++) {
      const int qrow = v + 8 * half;                 // row within the 16-q tile
      const float lrow = __shfl(lsum, qrow, 32);     // denominator of that query
      const float inv = 1.0f / lrow;
      if (stream == 0) {
#pragma unroll
        for (int j = 0; j < 8; j++)
          buf[qrow * HEAD_DIM + j * 16 + l16] = acc[j][v] * inv;
      } else {
        const size_t orow = ((size_t)(b * SEQ + (q0 + qrow))) * D_MODEL + h * HEAD_DIM;
#pragma unroll
        for (int j = 0; j < 8; j++) {
          const float fin = buf[qrow * HEAD_DIM + j * 16 + l16] - lam * acc[j][v] * inv;
          out[orow + j * 16 + l16] = (bf16_t)fin;
        }
      }
    }
    __syncthreads();
  }
}

// ---------------- host-side orchestration ----------------
extern "C" void kernel_launch(void* const* d_in, const int* in_sizes, int n_in,
                              void* d_out, int out_size, void* d_ws, size_t ws_size,
                              hipStream_t stream) {
  (void)in_sizes; (void)n_in; (void)out_size; (void)ws_size;
  const float* x = (const float*)d_in[0];
  const float* w_in[6] = { (const float*)d_in[1], (const float*)d_in[2], (const float*)d_in[3],
                           (const float*)d_in[4], (const float*)d_in[5], (const float*)d_in[6] };
  const float* lam = (const float*)d_in[7];
  float* out = (float*)d_out;

  uint8_t* p = (uint8_t*)d_ws;
  const size_t SZ_X = (size_t)MTOT * D_MODEL * sizeof(bf16_t);     // 16 MB
  const size_t SZ_W = (size_t)D_MODEL * D_MODEL * sizeof(bf16_t);  //  8 MB
  bf16_t* xb = (bf16_t*)p; p += SZ_X;
  bf16_t* wb[6];
  for (int i = 0; i < 6; i++) { wb[i] = (bf16_t*)p; p += SZ_W; }
  bf16_t* q1h = (bf16_t*)p; p += SZ_X;
  bf16_t* k1h = (bf16_t*)p; p += SZ_X;
  bf16_t* q2h = (bf16_t*)p; p += SZ_X;
  bf16_t* k2h = (bf16_t*)p; p += SZ_X;
  bf16_t* vTb = (bf16_t*)p; p += SZ_X;
  bf16_t* ao  = (bf16_t*)p; p += SZ_X;

  cvt_f32_bf16<<<2048, 256, 0, stream>>>(x, xb, MTOT * D_MODEL);
  for (int i = 0; i < 6; i++)
    cvt_f32_bf16<<<2048, 256, 0, stream>>>(w_in[i], wb[i], D_MODEL * D_MODEL);

  dim3 gg(D_MODEL / 128, MTOT / 128);  // (16, 32)
  gemm_bf16<<<gg, 256, 0, stream>>>(xb, wb[0], q1h, nullptr, MTOT, D_MODEL, D_MODEL, 1);
  gemm_bf16<<<gg, 256, 0, stream>>>(xb, wb[1], k1h, nullptr, MTOT, D_MODEL, D_MODEL, 1);
  gemm_bf16<<<gg, 256, 0, stream>>>(xb, wb[2], q2h, nullptr, MTOT, D_MODEL, D_MODEL, 1);
  gemm_bf16<<<gg, 256, 0, stream>>>(xb, wb[3], k2h, nullptr, MTOT, D_MODEL, D_MODEL, 1);
  gemm_bf16<<<gg, 256, 0, stream>>>(xb, wb[4], vTb, nullptr, MTOT, D_MODEL, D_MODEL, 2);

  dim3 ga(BATCH * NUM_HEADS, SEQ / 128);  // (32, 16)
  diff_attn<<<ga, 256, 0, stream>>>(q1h, k1h, q2h, k2h, vTb, lam, ao);

  gemm_bf16<<<gg, 256, 0, stream>>>(ao, wb[5], nullptr, out, MTOT, D_MODEL, D_MODEL, 0);
}
